// Net_56513179681389
// MI455X (gfx1250) — compile-verified, matrix operand order checked by reading, not run on hardware
//
#include <hip/hip_runtime.h>
#include <hip/hip_bf16.h>

typedef __bf16 bf16_t;
typedef __attribute__((ext_vector_type(16))) __bf16 v16bf;
typedef __attribute__((ext_vector_type(8)))  __bf16 v8bf;
typedef __attribute__((ext_vector_type(8)))  float  v8f;

#define DIM   300
#define KP    320          // padded K (10 tiles of 32)
#define KT    (KP / 32)
#define TOPK  10
#define WAVES 4            // waves per block; block = 128 threads
#define TILE_BYTES (16 * KP * 2)   // one 16-col B panel: 10240 B
#define CHUNKS     (TILE_BYTES / 16)   // 640 x 16B
#define CHUNKS_PER_THREAD (CHUNKS / 128) // 5

union FragU { v16bf v; struct { v8bf lo; v8bf hi; } h; };

// A-operand fragment from global (ISA 7.12.2 16-bit A layout):
// lane 0..15 -> row=lane, kb=0 ; lane 16..31 -> row=lane-16, kb=8
// elements: K in {kb..kb+7} U {kb+16..kb+23}
__device__ __forceinline__ v16bf load_frag_g(const bf16_t* __restrict__ base, int lane) {
    int r  = lane & 15;
    int kb = (lane >> 4) << 3;
    const bf16_t* p = base + r * KP + kb;
    FragU f;
    f.h.lo = *(const v8bf*)(p);
    f.h.hi = *(const v8bf*)(p + 16);
    return f.v;
}

// B-operand fragment from the LDS-staged 16xKP panel (row-major, row stride KP)
__device__ __forceinline__ v16bf load_frag_lds(const bf16_t* buf, int kt, int lane) {
    int c  = lane & 15;
    int kb = (lane >> 4) << 3;
    const bf16_t* p = buf + c * KP + kt * 32 + kb;
    FragU f;
    f.h.lo = *(const v8bf*)(p);
    f.h.hi = *(const v8bf*)(p + 16);
    return f.v;
}

// Async-stage one 16-column B panel (tile ct) into LDS. 5 x b128 per thread.
// Tracked by ASYNCcnt; caller pairs with s_wait_asynccnt + barrier.
__device__ __forceinline__ void stage_tile(const bf16_t* __restrict__ B, int ct,
                                           bf16_t* dst, int tid) {
    const char* gsrc = (const char*)(B + (size_t)ct * 16 * KP);
    unsigned ldsbase = (unsigned)(size_t)dst;      // ISA 10.2: low 32 bits of the pointer give the LDS address
#pragma unroll
    for (int j = 0; j < CHUNKS_PER_THREAD; ++j) {
        int c = tid + 128 * j;
        unsigned loff = ldsbase + c * 16;
        const void* g = gsrc + (size_t)c * 16;
        asm volatile("global_load_async_to_lds_b128 %0, %1, off"
                     :: "v"(loff), "v"(g) : "memory");
    }
}

// ---------------- pre-pass: src = emb_src @ W^T, row-normalize, -> bf16 padded ----
__global__ void __launch_bounds__(KP)
map_norm_src_kernel(const float* __restrict__ emb, const float* __restrict__ W,
                    bf16_t* __restrict__ outbf) {
    __shared__ float s_e[DIM];
    __shared__ float s_red[KP];
    const int i = blockIdx.x;
    for (int k = threadIdx.x; k < DIM; k += blockDim.x) s_e[k] = emb[(size_t)i * DIM + k];
    __syncthreads();
    const int d = threadIdx.x;
    float acc = 0.f;
    if (d < DIM) {
        const float* wr = W + (size_t)d * DIM;
        for (int k = 0; k < DIM; ++k) acc = fmaf(s_e[k], wr[k], acc);
    }
    s_red[d] = acc * acc;
    __syncthreads();
    if (d < 64) s_red[d] += s_red[d + 256];
    __syncthreads();
    for (int s = 128; s >= 1; s >>= 1) {
        if (d < s) s_red[d] += s_red[d + s];
        __syncthreads();
    }
    const float inv = rsqrtf(s_red[0] + 1e-30f);
    outbf[(size_t)i * KP + d] = (d < DIM) ? (bf16_t)(acc * inv) : (bf16_t)0.0f;
}

__global__ void __launch_bounds__(KP)
norm_tgt_kernel(const float* __restrict__ emb, bf16_t* __restrict__ outbf) {
    __shared__ float s_red[KP];
    const int i = blockIdx.x;
    const int d = threadIdx.x;
    float v = (d < DIM) ? emb[(size_t)i * DIM + d] : 0.f;
    s_red[d] = v * v;
    __syncthreads();
    if (d < 64) s_red[d] += s_red[d + 256];
    __syncthreads();
    for (int s = 128; s >= 1; s >>= 1) {
        if (d < s) s_red[d] += s_red[d + s];
        __syncthreads();
    }
    const float inv = rsqrtf(s_red[0] + 1e-30f);
    outbf[(size_t)i * KP + d] = (d < DIM) ? (bf16_t)(v * inv) : (bf16_t)0.0f;
}

// ---------------- WMMA pass 1: per-row top-10 mean of A @ B^T -----------------
// Block: 4 waves = 4 distinct row tiles; all waves share LDS-staged column tiles.
__global__ void __launch_bounds__(32 * WAVES)
sims_rowtop10_kernel(const bf16_t* __restrict__ A, const bf16_t* __restrict__ B,
                     float* __restrict__ avg_out, int N) {
    __shared__ bf16_t s_B[2][16 * KP];                 // double-buffered B panel
    __shared__ float  s_top[WAVES][16][TOPK];
    const int tid  = threadIdx.x;
    const int lane = tid & 31;
    const int wv   = tid >> 5;
    const int nRowT = N >> 4, nColT = N >> 4;
    int rt = blockIdx.x * WAVES + wv;
    if (rt >= nRowT) rt = nRowT - 1;                   // clamp: duplicate-but-identical work
    const int row0 = rt * 16;

    for (int idx = tid; idx < WAVES * 16 * TOPK; idx += blockDim.x)
        ((float*)s_top)[idx] = -3.0e38f;

    v16bf afrag[KT];
#pragma unroll
    for (int kt = 0; kt < KT; ++kt)
        afrag[kt] = load_frag_g(A + (size_t)row0 * KP + kt * 32, lane);

    const int mhalf = (lane >> 4) << 3;

    stage_tile(B, 0, &s_B[0][0], tid);                 // prologue prefetch

    for (int ct = 0; ct < nColT; ++ct) {
        const int cur = ct & 1;
        if (ct + 1 < nColT) {
            stage_tile(B, ct + 1, &s_B[cur ^ 1][0], tid);
            asm volatile("s_wait_asynccnt 0x5" ::: "memory");  // current panel done
        } else {
            asm volatile("s_wait_asynccnt 0x0" ::: "memory");
        }
        __syncthreads();                                // all waves' staging visible

        v16bf bfrag[KT];
#pragma unroll
        for (int kt = 0; kt < KT; ++kt)
            bfrag[kt] = load_frag_lds(&s_B[cur][0], kt, lane);
        v8f acc = {};
#pragma unroll
        for (int kt = 0; kt < KT; ++kt)
            acc = __builtin_amdgcn_wmma_f32_16x16x32_bf16(
                false, afrag[kt], false, bfrag[kt], (short)0, acc, false, false);

#pragma unroll
        for (int r = 0; r < 8; ++r) {
            float v = acc[r];
            int row = r + mhalf;
            bool want = v > s_top[wv][row][TOPK - 1];
            unsigned long long msk = __ballot(want);
            while (msk) {
                int leader = __ffsll(msk) - 1;
                if (lane == leader) {
                    float* a = s_top[wv][row];
                    if (v > a[TOPK - 1]) {
                        int j = TOPK - 1;
                        while (j > 0 && a[j - 1] < v) { a[j] = a[j - 1]; --j; }
                        a[j] = v;
                    }
                }
                msk &= msk - 1;
            }
        }
        __syncthreads();                                // panel free before restage
    }

    if (lane < 16) {                                    // each wave owns its 16 rows
        float sum = 0.f;
#pragma unroll
        for (int k = 0; k < TOPK; ++k) sum += s_top[wv][lane][k];
        avg_out[row0 + lane] = sum * (1.0f / TOPK);
    }
}

// ---------------- WMMA pass 2: per-row CSLS top-2 (+argmax of top-1) ----------
__global__ void __launch_bounds__(32 * WAVES)
csls_top2_kernel(const bf16_t* __restrict__ A, const bf16_t* __restrict__ B,
                 const float* __restrict__ avgS, const float* __restrict__ avgT,
                 float* __restrict__ b1, float* __restrict__ b2,
                 int* __restrict__ i1o, int N) {
    __shared__ bf16_t s_B[2][16 * KP];
    __shared__ float  s_v1[WAVES][16], s_v2[WAVES][16];
    __shared__ int    s_i1[WAVES][16];
    __shared__ float  s_as[WAVES][16];
    const int tid  = threadIdx.x;
    const int lane = tid & 31;
    const int wv   = tid >> 5;
    const int nRowT = N >> 4, nColT = N >> 4;
    int rt = blockIdx.x * WAVES + wv;
    if (rt >= nRowT) rt = nRowT - 1;
    const int row0 = rt * 16;

    if (lane < 16) {
        s_as[wv][lane] = avgS[row0 + lane];
        s_v1[wv][lane] = -3.0e38f;
        s_v2[wv][lane] = -3.0e38f;
        s_i1[wv][lane] = 0;
    }

    v16bf afrag[KT];
#pragma unroll
    for (int kt = 0; kt < KT; ++kt)
        afrag[kt] = load_frag_g(A + (size_t)row0 * KP + kt * 32, lane);

    const int mhalf = (lane >> 4) << 3;

    stage_tile(B, 0, &s_B[0][0], tid);

    for (int ct = 0; ct < nColT; ++ct) {
        const int cur = ct & 1;
        const int col = ct * 16 + (lane & 15);
        const float at = avgT[col];
        if (ct + 1 < nColT) {
            stage_tile(B, ct + 1, &s_B[cur ^ 1][0], tid);
            asm volatile("s_wait_asynccnt 0x5" ::: "memory");
        } else {
            asm volatile("s_wait_asynccnt 0x0" ::: "memory");
        }
        __syncthreads();

        v16bf bfrag[KT];
#pragma unroll
        for (int kt = 0; kt < KT; ++kt)
            bfrag[kt] = load_frag_lds(&s_B[cur][0], kt, lane);
        v8f acc = {};
#pragma unroll
        for (int kt = 0; kt < KT; ++kt)
            acc = __builtin_amdgcn_wmma_f32_16x16x32_bf16(
                false, afrag[kt], false, bfrag[kt], (short)0, acc, false, false);

#pragma unroll
        for (int r = 0; r < 8; ++r) {
            int row = r + mhalf;
            float v = 2.0f * acc[r] - s_as[wv][row] - at;
            bool want = v > s_v2[wv][row];
            unsigned long long msk = __ballot(want);
            while (msk) {
                int leader = __ffsll(msk) - 1;
                if (lane == leader) {
                    if (v > s_v1[wv][row]) {
                        s_v2[wv][row] = s_v1[wv][row];
                        s_v1[wv][row] = v;
                        s_i1[wv][row] = col;
                    } else if (v > s_v2[wv][row]) {
                        s_v2[wv][row] = v;
                    }
                }
                msk &= msk - 1;
            }
        }
        __syncthreads();
    }

    if (lane < 16) {
        b1[row0 + lane]  = s_v1[wv][lane];
        b2[row0 + lane]  = s_v2[wv][lane];
        i1o[row0 + lane] = s_i1[wv][lane];
    }
}

// ---------------- exact stable descending rank of diff = b1 - b2 --------------
__global__ void __launch_bounds__(256)
rank_kernel(const float* __restrict__ b1, const float* __restrict__ b2,
            int* __restrict__ order, int N) {
    __shared__ int s_cnt[256];
    const int i  = blockIdx.x;
    const float di = b1[i] - b2[i];
    int cnt = 0;
    for (int j = threadIdx.x; j < N; j += blockDim.x) {
        float dj = b1[j] - b2[j];
        if (dj > di || (dj == di && j < i)) ++cnt;
    }
    s_cnt[threadIdx.x] = cnt;
    __syncthreads();
    for (int s = 128; s >= 1; s >>= 1) {
        if (threadIdx.x < s) s_cnt[threadIdx.x] += s_cnt[threadIdx.x + s];
        __syncthreads();
    }
    if (threadIdx.x == 0) order[s_cnt[0]] = i;
}

// ---------------- emit dico/scores + pair cosines (deterministic) -------------
__global__ void __launch_bounds__(64)
finalize_kernel(const bf16_t* __restrict__ Sbf, const bf16_t* __restrict__ Tbf,
                const int* __restrict__ order, const int* __restrict__ i1,
                const float* __restrict__ b1, float* __restrict__ out,
                float* __restrict__ pcos, int N) {
    __shared__ float sr[64];
    const int p = blockIdx.x;
    const int i = order[p];
    const int j = i1[i];
    float acc = 0.f;
    for (int d = threadIdx.x; d < KP; d += blockDim.x)
        acc = fmaf((float)Sbf[(size_t)i * KP + d], (float)Tbf[(size_t)j * KP + d], acc);
    sr[threadIdx.x] = acc;
    __syncthreads();
    for (int s = 32; s >= 1; s >>= 1) {
        if (threadIdx.x < s) sr[threadIdx.x] += sr[threadIdx.x + s];
        __syncthreads();
    }
    if (threadIdx.x == 0) {
        pcos[p] = sr[0];
        int* oi = (int*)(out + 1);           // dico region: [1 .. 1+2N) as ints
        oi[2 * p]     = i;
        oi[2 * p + 1] = j;
        out[1 + 2 * (size_t)N + p] = b1[i];  // scores region
    }
}

__global__ void __launch_bounds__(256)
mean_kernel(const float* __restrict__ pcos, float* __restrict__ out, int N) {
    __shared__ float sr[256];
    float acc = 0.f;
    for (int p = threadIdx.x; p < N; p += 256) acc += pcos[p];
    sr[threadIdx.x] = acc;
    __syncthreads();
    for (int s = 128; s >= 1; s >>= 1) {
        if (threadIdx.x < s) sr[threadIdx.x] += sr[threadIdx.x + s];
        __syncthreads();
    }
    if (threadIdx.x == 0) out[0] = sr[0] / (float)N;
}

extern "C" void kernel_launch(void* const* d_in, const int* in_sizes, int n_in,
                              void* d_out, int out_size, void* d_ws, size_t ws_size,
                              hipStream_t stream) {
    const float* emb_src = (const float*)d_in[0];
    const float* emb_tgt = (const float*)d_in[1];
    const float* W       = (const float*)d_in[2];
    const int N = in_sizes[0] / DIM;            // 10000, multiple of 16

    char* ws = (char*)d_ws;
    size_t off = 0;
    bf16_t* Sbf  = (bf16_t*)(ws + off); off += (size_t)N * KP * sizeof(bf16_t);
    bf16_t* Tbf  = (bf16_t*)(ws + off); off += (size_t)N * KP * sizeof(bf16_t);
    float*  avgS = (float*) (ws + off); off += (size_t)N * sizeof(float);
    float*  avgT = (float*) (ws + off); off += (size_t)N * sizeof(float);
    float*  b1   = (float*) (ws + off); off += (size_t)N * sizeof(float);
    float*  b2   = (float*) (ws + off); off += (size_t)N * sizeof(float);
    int*    i1   = (int*)   (ws + off); off += (size_t)N * sizeof(int);
    int*    ord  = (int*)   (ws + off); off += (size_t)N * sizeof(int);
    float*  pcos = (float*) (ws + off); off += (size_t)N * sizeof(float);

    float* out = (float*)d_out;

    const int nRowT = N / 16;
    const int nBlk  = (nRowT + WAVES - 1) / WAVES;

    map_norm_src_kernel<<<N, KP, 0, stream>>>(emb_src, W, Sbf);
    norm_tgt_kernel    <<<N, KP, 0, stream>>>(emb_tgt, Tbf);

    sims_rowtop10_kernel<<<nBlk, 32 * WAVES, 0, stream>>>(Sbf, Tbf, avgS, N); // r_S
    sims_rowtop10_kernel<<<nBlk, 32 * WAVES, 0, stream>>>(Tbf, Sbf, avgT, N); // r_T

    csls_top2_kernel<<<nBlk, 32 * WAVES, 0, stream>>>(Sbf, Tbf, avgS, avgT, b1, b2, i1, N);

    rank_kernel<<<N, 256, 0, stream>>>(b1, b2, ord, N);
    finalize_kernel<<<N, 64, 0, stream>>>(Sbf, Tbf, ord, i1, b1, out, pcos, N);
    mean_kernel<<<1, 256, 0, stream>>>(pcos, out, N);
}
